// MemoryGate_12017318494276
// MI455X (gfx1250) — compile-verified, matrix operand order checked by reading, not run on hardware
//
#include <hip/hip_runtime.h>
#include <hip/hip_bf16.h>

#define B_ 8
#define N_ 512
#define T_ 64
#define C_ 128
#define M_ 32
#define MEM_ 20
#define IN_ 2
#define E_ 4
#define G_ 4     // n-tiles per block (weights amortized over G_)

typedef __attribute__((ext_vector_type(16))) __bf16 v16bf;
typedef __attribute__((ext_vector_type(8)))  float  v8f;
typedef __attribute__((ext_vector_type(4)))  float  v4f;
typedef __attribute__((ext_vector_type(4)))  unsigned int v4u;
typedef __attribute__((ext_vector_type(2)))  unsigned int v2u;

// ---- Fused-kernel LDS layout (bytes); fragment-major operand tiles (512 bf16,
//      lane l's 16 values contiguous at tile + l*16).
#define WTS_OFF  0          // bf16 [p3][Kt4][Nt2][512] = 24576 (persistent)
#define HID_OFF  24576      // hidA 16384 | en f32 [64][66]=16896 | att f32 [64][33]=8448
#define EN_OFF   24576
#define ATT_OFF  24576
#define QA_OFF   41472      // 4096
#define KF_OFF   45568      // 4096   (PsA [4][2][512]=8192 overlays qA+kF)
#define PSA_OFF  41472
#define VF_OFF   49664      // 4096
#define MEMS_OFF 53760      // f32 [64][33] = 8448 (fallback only)
#define SMEM_BYTES 62208

#define EN_STR 66
#define AT_STR 33

// Load one fragment (16 bf16 per lane, contiguous) -> 2x ds_load_b128
__device__ __forceinline__ v16bf load_frag(const __bf16* tile, int lane) {
    const v4u* p = (const v4u*)tile + (lane << 1);
    union { v16bf v; v4u u[2]; } U;
    U.u[0] = p[0];
    U.u[1] = p[1];
    return U.v;
}

// Pack 4 floats -> 4 bf16 -> single 8B LDS store (dst must be 8B aligned)
__device__ __forceinline__ void store4bf(__bf16* dst, v4f f) {
    union { __bf16 h[4]; v2u u; } U;
    U.h[0] = (__bf16)f[0]; U.h[1] = (__bf16)f[1];
    U.h[2] = (__bf16)f[2]; U.h[3] = (__bf16)f[3];
    *(v2u*)dst = U.u;
}

// D fragment (16x16 f32): lane&15 = col N; VGPR r holds row r (lanes 0-15) / 8+r (lanes 16-31)
__device__ __forceinline__ void store_D_f32(float* lds, int stride, int row0, int col0, int lane, v8f d) {
    int nn = lane & 15;
    int mb = ((lane >> 4) & 1) * 8;
#pragma unroll
    for (int r = 0; r < 8; ++r) lds[(row0 + mb + r) * stride + col0 + nn] = d[r];
}

// ---- Kernel 1: memory-bank soft read on WMMA.
//      logits = q_in[rows,32] @ mem^T[32,20->pad32]; read = softmax(logits) @ mem[20->pad32,32]
#define MR_ROWS 128
__global__ __launch_bounds__(128)
void memread_wmma_kernel(const float* __restrict__ input,
                         const float* __restrict__ mem,
                         const float* __restrict__ iq,
                         float* __restrict__ ws)
{
    __shared__ __align__(16) char sm[8192 + 16896 + 4096];
    __bf16* rowA = (__bf16*)(sm);               // [8][512] A-frags: q_in rows, reused for P
    float*  lg   = (float*)(sm + 8192);         // [128][33] logits (f32)
    __bf16* fL   = (__bf16*)(sm + 8192 + 16896);// [2][512] B-frags: K=m, N=k (logits)
    __bf16* fC   = fL + 1024;                   // [2][512] B-frags: K=k, N=m (combine)
    const int tid = threadIdx.x, lane = tid & 31, wave = tid >> 5;
    const size_t row0 = (size_t)blockIdx.x * MR_ROWS;

    // zero-fill both mem fragments (covers the k>=20 / padded slots)
    {
        v4u z = {};
        ((v4u*)fL)[tid] = z;    // 128*16B = 2048B
        ((v4u*)fC)[tid] = z;
    }
    __syncthreads();
    // fill valid mem entries into both fragment layouts (640 elems, 5 per thread)
#pragma unroll
    for (int ii = 0; ii < 5; ++ii) {
        int idx = tid * 5 + ii;                 // 0..639
        int k = idx >> 5, m = idx & 31;
        __bf16 hv = (__bf16)mem[idx];
        fL[(k >> 4) * 512 + (((m >> 3) & 1) * 16 + (k & 15)) * 16 + (m & 7) + ((m >> 4) & 1) * 8] = hv;
        fC[(m >> 4) * 512 + (((k >> 3) & 1) * 16 + (m & 15)) * 16 + (k & 7) + ((k >> 4) & 1) * 8] = hv;
    }
    // q_in = input @ iq, written straight into A-fragment layout (4x 16B stores)
    float in0 = input[(row0 + tid) * IN_ + 0];
    float in1 = input[(row0 + tid) * IN_ + 1];
    const int t16 = tid >> 4, rl = tid & 15;
    __bf16* fr = rowA + t16 * 512;
#pragma unroll
    for (int kh = 0; kh < 2; ++kh)
#pragma unroll
        for (int jh = 0; jh < 2; ++jh) {
            union { __bf16 h[8]; v4u u; } U;
#pragma unroll
            for (int j = 0; j < 8; ++j) {
                int m = jh * 16 + kh * 8 + j;
                U.h[j] = (__bf16)(in0 * iq[m] + in1 * iq[M_ + m]);
            }
            *(v4u*)(fr + (kh * 16 + rl) * 16 + jh * 8) = U.u;
        }
    __syncthreads();

    // logits: 8 row-tiles x 2 N-tiles, 1 WMMA each
#pragma unroll 1
    for (int t = wave; t < 8; t += 4) {
        v16bf a = load_frag(rowA + t * 512, lane);
#pragma unroll
        for (int Nt = 0; Nt < 2; ++Nt) {
            v16bf bb = load_frag(fL + Nt * 512, lane);
            v8f d = {};
            d = __builtin_amdgcn_wmma_f32_16x16x32_bf16(false, a, false, bb,
                                                        (short)0, d, false, false);
            store_D_f32(lg, 33, t * 16, Nt * 16, lane, d);
        }
    }
    __syncthreads();

    // per-row softmax over 20 valid logits; write P into A-frag layout (zero-padded)
    {
        float* r = lg + tid * 33;
        float mx = -1e30f;
#pragma unroll
        for (int k = 0; k < MEM_; ++k) mx = fmaxf(mx, r[k]);
        float sum = 0.f;
        float pv[MEM_];
#pragma unroll
        for (int k = 0; k < MEM_; ++k) { pv[k] = __expf(r[k] - mx); sum += pv[k]; }
        float inv = 1.f / sum;
#pragma unroll
        for (int kh = 0; kh < 2; ++kh)
#pragma unroll
            for (int jh = 0; jh < 2; ++jh) {
                union { __bf16 h[8]; v4u u; } U;
#pragma unroll
                for (int j = 0; j < 8; ++j) {
                    int k = jh * 16 + kh * 8 + j;
                    U.h[j] = (__bf16)((k < MEM_) ? pv[k] * inv : 0.f);
                }
                *(v4u*)(fr + (kh * 16 + rl) * 16 + jh * 8) = U.u;
            }
    }
    __syncthreads();

    // combine: read = P @ mem, store straight to workspace
#pragma unroll 1
    for (int t = wave; t < 8; t += 4) {
        v16bf a = load_frag(rowA + t * 512, lane);
        int nn = lane & 15, mb = ((lane >> 4) & 1) * 8;
#pragma unroll
        for (int Nt = 0; Nt < 2; ++Nt) {
            v16bf bb = load_frag(fC + Nt * 512, lane);
            v8f d = {};
            d = __builtin_amdgcn_wmma_f32_16x16x32_bf16(false, a, false, bb,
                                                        (short)0, d, false, false);
#pragma unroll
            for (int rr = 0; rr < 8; ++rr)
                ws[(row0 + t * 16 + mb + rr) * M_ + Nt * 16 + nn] = d[rr];
        }
    }
}

// ---- Kernel 2: fused qkv projection + attention + cosine gate ----
__global__ __launch_bounds__(128)
void memgate_fused_kernel(const float* __restrict__ input,
                          const float* __restrict__ hidden,
                          const float* __restrict__ mem,
                          const float* __restrict__ iq,
                          const float* __restrict__ Wq,
                          const float* __restrict__ Wk,
                          const float* __restrict__ Wv,
                          float* __restrict__ out,
                          const float* __restrict__ memsG,
                          int use_ws)
{
    __shared__ __align__(16) char smem[SMEM_BYTES];
    __bf16* wtsF = (__bf16*)(smem + WTS_OFF);
    __bf16* hidA = (__bf16*)(smem + HID_OFF);
    float*  en   = (float*)(smem + EN_OFF);
    float*  att  = (float*)(smem + ATT_OFF);
    __bf16* qA   = (__bf16*)(smem + QA_OFF);
    __bf16* kF   = (__bf16*)(smem + KF_OFF);
    __bf16* PsA  = (__bf16*)(smem + PSA_OFF);
    __bf16* vF   = (__bf16*)(smem + VF_OFF);
    float*  mems = (float*)(smem + MEMS_OFF);

    const int n0 = blockIdx.x * G_, b = blockIdx.y, e = blockIdx.z;
    const int tid  = threadIdx.x;
    const int lane = tid & 31;
    const int wave = tid >> 5;

    // ---- Stage Wq/Wk/Wv[e] once: [128][32] f32 -> bf16 B-fragment layout ----
    {
        const v4f* w4[3] = { (const v4f*)(Wq + (size_t)e * C_ * M_),
                             (const v4f*)(Wk + (size_t)e * C_ * M_),
                             (const v4f*)(Wv + (size_t)e * C_ * M_) };
#pragma unroll
        for (int p = 0; p < 3; ++p) {
#pragma unroll
            for (int i = 0; i < 8; ++i) {
                int idx4 = tid + i * 128;
                v4f f = w4[p][idx4];
                int k    = idx4 >> 3;
                int m0   = (idx4 & 7) * 4;
                int Kt   = k >> 5;
                int krem = k & 31;
                int khB  = (krem >> 3) & 1;
                int jB   = (krem & 7) + ((krem >> 4) & 1) * 8;
                int Nt   = m0 >> 4;
                int nn0  = m0 & 15;
                __bf16* base = wtsF + ((p * 4 + Kt) * 2 + Nt) * 512 + jB;
                base[(khB * 16 + nn0 + 0) * 16] = (__bf16)f[0];
                base[(khB * 16 + nn0 + 1) * 16] = (__bf16)f[1];
                base[(khB * 16 + nn0 + 2) * 16] = (__bf16)f[2];
                base[(khB * 16 + nn0 + 3) * 16] = (__bf16)f[3];
            }
        }
    }

    const v4f* hsrc4 = (const v4f*)(hidden + ((size_t)((e * B_ + b) * N_ + n0)) * (T_ * C_));

#pragma unroll 1
    for (int g = 0; g < G_; ++g) {
        const int n = n0 + g;
        __syncthreads();   // protect att/mems/hidA region from previous iteration readers

        // ---- stage hidden tile (non-temporal: streamed once grid-wide) -> A-frag LDS ----
        {
            const v4f* src = hsrc4 + (size_t)g * (T_ * C_ / 4);
#pragma unroll
            for (int i = 0; i < 16; ++i) {
                int idx4 = tid + i * 128;
                v4f f = __builtin_nontemporal_load(src + idx4);
                int t    = idx4 >> 5;
                int c0   = (idx4 & 31) * 4;
                int Mt   = t >> 4;
                int Kt   = c0 >> 5;
                int krem = c0 & 31;
                int kh   = (krem >> 3) & 1;
                int j0   = (krem & 7) + ((krem >> 4) & 1) * 8;
                store4bf(hidA + (Mt * 4 + Kt) * 512 + (kh * 16 + (t & 15)) * 16 + j0, f);
            }
        }
        // ---- prefetch next tile into cache (global_prefetch, no VGPR cost) ----
        if (g + 1 < G_) {
            const char* nx = (const char*)(hsrc4 + (size_t)(g + 1) * (T_ * C_ / 4));
            __builtin_prefetch(nx + tid * 256, 0, 0);
            __builtin_prefetch(nx + tid * 256 + 128, 0, 0);
        }

        // ---- compact inline fallback (cold path; logits recomputed, k-loops rolled) ----
        if (!use_ws && tid < T_) {
            const float* isrc = input + ((size_t)((b * N_ + n) * T_ + tid)) * IN_;
            float in0 = isrc[0], in1 = isrc[1];
            float qv[M_];
#pragma unroll
            for (int m = 0; m < M_; ++m) qv[m] = in0 * iq[m] + in1 * iq[M_ + m];
            float mx = -1e30f;
#pragma unroll 1
            for (int k = 0; k < MEM_; ++k) {
                float s = 0.f;
#pragma unroll
                for (int m = 0; m < M_; ++m) s += qv[m] * mem[k * M_ + m];
                mx = fmaxf(mx, s);
            }
            float sum = 0.f;
#pragma unroll 1
            for (int k = 0; k < MEM_; ++k) {
                float s = 0.f;
#pragma unroll
                for (int m = 0; m < M_; ++m) s += qv[m] * mem[k * M_ + m];
                sum += __expf(s - mx);
            }
            float inv = 1.f / sum;
            float acc[M_] = {};
#pragma unroll 1
            for (int k = 0; k < MEM_; ++k) {
                float s = 0.f;
#pragma unroll
                for (int m = 0; m < M_; ++m) s += qv[m] * mem[k * M_ + m];
                float w = __expf(s - mx);
#pragma unroll
                for (int m = 0; m < M_; ++m) acc[m] += w * mem[k * M_ + m];
            }
#pragma unroll
            for (int m = 0; m < M_; ++m) mems[tid * AT_STR + m] = acc[m] * inv;
        }
        __syncthreads();

        // ---- q/k/v projections: [64x128]@[128x32], 24 tiles over 4 waves ----
#pragma unroll 1
        for (int job = wave; job < 24; job += 4) {
            int p  = job >> 3;
            int tl = job & 7;
            int Mt = tl >> 1, Nt = tl & 1;
            v8f acc = {};
#pragma unroll
            for (int Kt = 0; Kt < 4; ++Kt) {
                v16bf a  = load_frag(hidA + (Mt * 4 + Kt) * 512, lane);
                v16bf bb = load_frag(wtsF + ((p * 4 + Kt) * 2 + Nt) * 512, lane);
                acc = __builtin_amdgcn_wmma_f32_16x16x32_bf16(false, a, false, bb,
                                                              (short)0, acc, false, false);
            }
            int nn  = lane & 15;
            int mb  = ((lane >> 4) & 1) * 8;
            int khm = (nn >> 3) & 1;
            int jm  = (nn & 7) + Nt * 8;
            if (p == 0) {
                __bf16* base = qA + Mt * 512 + jm;
#pragma unroll
                for (int r = 0; r < 8; ++r) base[(khm * 16 + mb + r) * 16] = (__bf16)acc[r];
            } else if (p == 1) {
                __bf16* base = kF + Mt * 512 + jm;
#pragma unroll
                for (int r = 0; r < 8; ++r) base[(khm * 16 + mb + r) * 16] = (__bf16)acc[r];
            } else {
                int f    = (Mt >> 1) * 2 + Nt;
                int s_hi = (Mt & 1) * 8;
                __bf16* base = vF + f * 512;
#pragma unroll
                for (int r = 0; r < 8; ++r) {
                    int sr  = mb + r;
                    int khs = (sr >> 3) & 1;
                    int js  = (sr & 7) + s_hi;
                    base[(khs * 16 + nn) * 16 + js] = (__bf16)acc[r];
                }
            }
        }
        __syncthreads();

        // ---- energy = q @ k^T : 16 tiles, 1 WMMA each ----
#pragma unroll 1
        for (int t = wave; t < 16; t += 4) {
            int Mt = t >> 2, Nt = t & 3;
            v16bf a  = load_frag(qA + Mt * 512, lane);
            v16bf bb = load_frag(kF + Nt * 512, lane);
            v8f acc = {};
            acc = __builtin_amdgcn_wmma_f32_16x16x32_bf16(false, a, false, bb,
                                                          (short)0, acc, false, false);
            store_D_f32(en, EN_STR, Mt * 16, Nt * 16, lane, acc);
        }
        __syncthreads();

        // ---- row softmax, emit P in A-fragment layout ----
        if (tid < T_) {
            float* row = en + tid * EN_STR;
            float mx = -1e30f;
#pragma unroll 8
            for (int s = 0; s < T_; ++s) mx = fmaxf(mx, row[s]);
            float sum = 0.f;
#pragma unroll 8
            for (int s = 0; s < T_; ++s) { float v = __expf(row[s] - mx); row[s] = v; sum += v; }
            float inv = 1.f / sum;
            int Mt = tid >> 4;
            int rl = tid & 15;
#pragma unroll
            for (int s4 = 0; s4 < T_; s4 += 4) {
                int Kt   = s4 >> 5;
                int krem = s4 & 31;
                int kh   = (krem >> 3) & 1;
                int j0   = (krem & 7) + ((krem >> 4) & 1) * 8;
                union { __bf16 h[4]; v2u u; } U;
                U.h[0] = (__bf16)(row[s4 + 0] * inv);
                U.h[1] = (__bf16)(row[s4 + 1] * inv);
                U.h[2] = (__bf16)(row[s4 + 2] * inv);
                U.h[3] = (__bf16)(row[s4 + 3] * inv);
                *(v2u*)(PsA + (Mt * 2 + Kt) * 512 + (kh * 16 + rl) * 16 + j0) = U.u;
            }
        }
        __syncthreads();

        // ---- att = P @ v : 8 tiles x 2 WMMAs ----
#pragma unroll 1
        for (int t = wave; t < 8; t += 4) {
            int Mt = t >> 1, Nt = t & 1;
            v8f acc = {};
#pragma unroll
            for (int Kt = 0; Kt < 2; ++Kt) {
                v16bf a  = load_frag(PsA + (Mt * 2 + Kt) * 512, lane);
                v16bf bb = load_frag(vF + (Kt * 2 + Nt) * 512, lane);
                acc = __builtin_amdgcn_wmma_f32_16x16x32_bf16(false, a, false, bb,
                                                              (short)0, acc, false, false);
            }
            store_D_f32(att, AT_STR, Mt * 16, Nt * 16, lane, acc);
        }
        __syncthreads();

        // ---- cosine similarity + output [B,N,T,1,E] ----
        if (tid < T_) {
            size_t row = (size_t)((b * N_ + n) * T_ + tid);
            const float* mrow = use_ws ? (memsG + row * M_) : (mems + tid * AT_STR);
            float dot = 0.f, na = 0.f, nb = 0.f;
#pragma unroll
            for (int m = 0; m < M_; ++m) {
                float x = mrow[m];
                float y = att[tid * AT_STR + m];
                dot += x * y; na += x * x; nb += y * y;
            }
            na = fmaxf(sqrtf(na), 1e-8f);
            nb = fmaxf(sqrtf(nb), 1e-8f);
            out[row * E_ + e] = dot / (na * nb);
        }
    }
}

extern "C" void kernel_launch(void* const* d_in, const int* in_sizes, int n_in,
                              void* d_out, int out_size, void* d_ws, size_t ws_size,
                              hipStream_t stream) {
    (void)in_sizes; (void)n_in; (void)out_size;
    const float* input  = (const float*)d_in[0];
    const float* hidden = (const float*)d_in[1];
    const float* mem    = (const float*)d_in[2];
    const float* iq     = (const float*)d_in[3];
    const float* Wq     = (const float*)d_in[4];
    const float* Wk     = (const float*)d_in[5];
    const float* Wv     = (const float*)d_in[6];
    float* out = (float*)d_out;

    const size_t need = (size_t)B_ * N_ * T_ * M_ * sizeof(float);
    const int use_ws = (d_ws != nullptr && ws_size >= need) ? 1 : 0;
    if (use_ws) {
        memread_wmma_kernel<<<(B_ * N_ * T_) / MR_ROWS, 128, 0, stream>>>(input, mem, iq, (float*)d_ws);
    }
    dim3 grid(N_ / G_, B_, E_);   // (n-tile, b, e)
    memgate_fused_kernel<<<grid, dim3(128), 0, stream>>>(input, hidden, mem, iq, Wq, Wk, Wv,
                                                         out, (const float*)d_ws, use_ws);
}
